// UpdateX_56358560858585
// MI455X (gfx1250) — matrix operand order repaired and checked by reading, not compiled, not run on hardware
//
#include <hip/hip_runtime.h>
#include <hip/hip_bf16.h>
#include <stddef.h>

// ---------------------------------------------------------------------------
// Shapes (from reference): B=1, R=128, L=256, IN_DIM=128, DIM_MSA=32, DIM=128
// out[i,j,e] = x[i,j,e] + b2[e] + (1/128) * sum_{r,c,d} mp[r,j,c]*mp[r,i,d]*W2[c*32+d, e]
//   with mp[r,l,c] = sum_d m[r,l,d]*W1[d,c] + b1[c]
// ---------------------------------------------------------------------------

typedef __attribute__((ext_vector_type(16))) _Float16 v16h;
typedef __attribute__((ext_vector_type(8)))  _Float16 v8h;
typedef __attribute__((ext_vector_type(8)))  float    v8f;

union FragH {
    v16h v;
    v8h  h8[2];
};

#define RR      128
#define LL      256
#define IN_DIM  128
#define DMSA    32
#define DIM     128
#define OP_PAD  1040   // 1024 + 16 halves padding for LDS bank spread

// ---------------------------------------------------------------------------
// Kernel 1: mp = m @ W1 + b1  (f32 accumulate), stored f16 in two layouts:
//   mpA [l][c][r]   : A-operand source (row-major 32x128 per l) -> contiguous B128 frags
//   mpBp[l][kk][dT][lane][h] : B-operand fragment layout, per-lane contiguous 16 halves
// One wave per (r,l) row, lane == c (0..31). 8 waves / block.
// ---------------------------------------------------------------------------
__global__ __launch_bounds__(256) void proj1_kernel(
    const float* __restrict__ m, const float* __restrict__ W1,
    const float* __restrict__ b1,
    _Float16* __restrict__ mpA, _Float16* __restrict__ mpBp)
{
    const int wave = threadIdx.x >> 5;
    const int c    = threadIdx.x & 31;
    const int row  = blockIdx.x * 8 + wave;   // row = r*256 + l, row < 32768
    const int r    = row >> 8;
    const int l    = row & 255;

    const float* mrow = m + (size_t)row * IN_DIM;
    float acc = b1[c];
#pragma unroll 4
    for (int d = 0; d < IN_DIM; ++d)
        acc = fmaf(mrow[d], W1[d * DMSA + c], acc);

    const _Float16 hv = (_Float16)acc;

    // mpA[l][c][r]
    mpA[((size_t)(l * DMSA + c)) * RR + r] = hv;

    // mpBp: K = h + 16*laneHi within chunk kk; col lane15 = c&15, dT = c>>4
    const int kk     = r >> 5;
    const int laneHi = (r >> 4) & 1;
    const int h      = r & 15;
    const int dT     = c >> 4;
    const int lane15 = c & 15;
    mpBp[((size_t)((l * 4 + kk) * 2 + dT)) * 512 + (size_t)(lane15 + 16 * laneHi) * 16 + h] = hv;
}

// ---------------------------------------------------------------------------
// Kernel 2: repack W2 (1024x128 f32) into B-fragment layout, f16:
//   w2p[kk(32)][nT(8)][lane(32)][h(16)] = W2[kk*32 + (h + 16*laneHi)][nT*16 + lane15]
// ---------------------------------------------------------------------------
__global__ __launch_bounds__(256) void w2prep_kernel(
    const float* __restrict__ W2, _Float16* __restrict__ w2p)
{
    const int tid = blockIdx.x * 256 + threadIdx.x;   // < 131072
    const int k = tid >> 7;
    const int e = tid & 127;
    const float v = W2[tid];

    const int kk     = k >> 5;
    const int klocal = k & 31;
    const int laneHi = klocal >> 4;
    const int h      = klocal & 15;
    const int nT     = e >> 4;
    const int lane15 = e & 15;
    w2p[((size_t)((kk * 8 + nT) * 32 + lane15 + 16 * laneHi)) * 16 + h] = (_Float16)v;
}

// ---------------------------------------------------------------------------
// Kernel 3 (fused): per block: i = blockIdx.y, j in [j0, j0+16).
//  Stage A: for each pair p (16) and 16x16 subtile (cT,dT):
//     op[c,d] = (1/128) sum_r mp[r,j,c] * mp[r,i,d]   via 4x wmma_f32_16x16x32_f16
//     -> f16 into LDS OP[p][c*32+d]
//  Stage B: wave w computes OUT[16 pairs][nT=w 16-wide] = OP(16x1024) x W2 + x + b2
// ---------------------------------------------------------------------------
__global__ __launch_bounds__(256) void fused_outer_proj_kernel(
    const float* __restrict__ x, const float* __restrict__ b2,
    const _Float16* __restrict__ mpA, const _Float16* __restrict__ mpBp,
    const _Float16* __restrict__ w2p, float* __restrict__ out)
{
    __shared__ _Float16 sOP[16 * OP_PAD];

    const int lane   = threadIdx.x & 31;
    const int wave   = threadIdx.x >> 5;
    const int laneHi = lane >> 4;
    const int lane15 = lane & 15;

    const int i  = blockIdx.y;
    const int j0 = blockIdx.x * 16;

    // -------------------- Stage A --------------------
    // 64 tile jobs: job -> pair p = job>>2, subtile s = job&3 (cT = s>>1, dT = s&1)
#pragma unroll
    for (int t = 0; t < 8; ++t) {
        const int job = wave * 8 + t;
        const int p   = job >> 2;
        const int cT  = (job >> 1) & 1;
        const int dT  = job & 1;
        const int j   = j0 + p;

        v8f acc = {};
        const _Float16* aRow = mpA + ((size_t)(j * DMSA + cT * 16 + lane15)) * RR;
#pragma unroll
        for (int kk = 0; kk < 4; ++kk) {
            FragH a, b;
            // A frag: lane<16 -> K 0..15 ; lane>=16 -> K 16..31 (two B128 loads)
            const _Float16* ap = aRow + kk * 32 + laneHi * 16;
            a.h8[0] = *(const v8h*)(ap);
            a.h8[1] = *(const v8h*)(ap + 8);
            // B frag: pre-packed, 16 contiguous halves per lane (two B128 loads)
            const _Float16* bp = mpBp + ((size_t)((i * 4 + kk) * 2 + dT)) * 512 + (size_t)lane * 16;
            b.h8[0] = *(const v8h*)(bp);
            b.h8[1] = *(const v8h*)(bp + 8);
            acc = __builtin_amdgcn_wmma_f32_16x16x32_f16(
                false, a.v, false, b.v, (short)0, acc, false, false);
        }
        // D layout: VGPR g -> row M = g + 8*laneHi, col N = lane15
        const int dcol = dT * 16 + lane15;
#pragma unroll
        for (int g = 0; g < 8; ++g) {
            const int crow = cT * 16 + g + 8 * laneHi;
            sOP[p * OP_PAD + crow * DMSA + dcol] = (_Float16)(acc[g] * 0.0078125f); // /128
        }
    }

    __syncthreads();

    // -------------------- Stage B --------------------
    const int nT = wave;       // 0..7 -> 16 output channels each
    v8f acc = {};
#pragma unroll 4
    for (int kk = 0; kk < 32; ++kk) {
        FragH a, b;
        // A frag rows = pairs (M = lane15), K pattern 0..15 (+16 for hi lanes)
        const _Float16* ap = &sOP[lane15 * OP_PAD + kk * 32 + laneHi * 16];
        a.h8[0] = *(const v8h*)(ap);
        a.h8[1] = *(const v8h*)(ap + 8);
        // W2 frag: pre-packed per-lane contiguous
        const _Float16* bp = w2p + ((size_t)(kk * 8 + nT)) * 512 + (size_t)lane * 16;
        b.h8[0] = *(const v8h*)(bp);
        b.h8[1] = *(const v8h*)(bp + 8);
        acc = __builtin_amdgcn_wmma_f32_16x16x32_f16(
            false, a.v, false, b.v, (short)0, acc, false, false);
    }

    // Epilogue: out[i, j0+p, e] = x + acc + b2[e]
    const int e = nT * 16 + lane15;
    const float bias = b2[e];
#pragma unroll
    for (int g = 0; g < 8; ++g) {
        const int p = g + 8 * laneHi;
        const size_t off = ((size_t)(i * LL + j0 + p)) * DIM + e;
        out[off] = x[off] + acc[g] + bias;
    }
}

// ---------------------------------------------------------------------------
extern "C" void kernel_launch(void* const* d_in, const int* in_sizes, int n_in,
                              void* d_out, int out_size, void* d_ws, size_t ws_size,
                              hipStream_t stream)
{
    (void)in_sizes; (void)n_in; (void)out_size; (void)ws_size;

    const float* x  = (const float*)d_in[0];   // [1,256,256,128]
    const float* m  = (const float*)d_in[1];   // [1,128,256,128]
    const float* W1 = (const float*)d_in[2];   // [128,32]
    const float* b1 = (const float*)d_in[3];   // [32]
    const float* W2 = (const float*)d_in[4];   // [1024,128]
    const float* b2 = (const float*)d_in[5];   // [128]
    float* out = (float*)d_out;                // [1,256,256,128]

    // Workspace carve-up (f16): mpA 1M, mpBp 1M, w2p 128K halves  (~4.25 MB)
    _Float16* mpA  = (_Float16*)d_ws;
    _Float16* mpBp = mpA  + (size_t)LL * DMSA * RR;          // 1,048,576
    _Float16* w2p  = mpBp + (size_t)LL * 4 * 2 * 512;        // 1,048,576

    // 1) mp projection + repack (32768 rows, 8 per block)
    proj1_kernel<<<dim3(4096), dim3(256), 0, stream>>>(m, W1, b1, mpA, mpBp);
    // 2) W2 repack (131072 elements)
    w2prep_kernel<<<dim3(512), dim3(256), 0, stream>>>(W2, w2p);
    // 3) fused outer-product-mean + projection + residual
    fused_outer_proj_kernel<<<dim3(16, 256), dim3(256), 0, stream>>>(
        x, b2, mpA, mpBp, w2p, out);
}
